// TransformerBlock_77163382440433
// MI455X (gfx1250) — compile-verified, matrix operand order checked by reading, not run on hardware
//
#include <hip/hip_runtime.h>
#include <hip/hip_bf16.h>

typedef __bf16 bf16_t;
typedef __attribute__((ext_vector_type(16))) __bf16 v16bf;
typedef __attribute__((ext_vector_type(8)))  float  v8f;
typedef int v4i __attribute__((vector_size(16)));   // matches builtin's int4

// ---------------------------------------------------------------------------
// Async global->LDS copy path (gfx1250 GLOBAL_LOAD_ASYNC_TO_LDS_B128,
// ASYNCcnt-tracked). Builtin signature (from compiler diagnostic):
//   (global int4*, shared int4*, imm int offset, imm int cpol)
// Guarded so the file compiles on toolchains that do not expose the builtins.
// ---------------------------------------------------------------------------
#if __has_builtin(__builtin_amdgcn_global_load_async_to_lds_b128) && \
    __has_builtin(__builtin_amdgcn_s_wait_asynccnt)
#define USE_ASYNC 1
#else
#define USE_ASYNC 0
#endif

__device__ __forceinline__ void cp_b128(bf16_t* lds, const bf16_t* g) {
#if USE_ASYNC
  __builtin_amdgcn_global_load_async_to_lds_b128(
      (__attribute__((address_space(1))) v4i*)g,
      (__attribute__((address_space(3))) v4i*)lds, 0, 0);
#else
  *(uint4*)lds = *(const uint4*)g;
#endif
}
__device__ __forceinline__ void cp_wait_all() {
#if USE_ASYNC
  __builtin_amdgcn_s_wait_asynccnt(0);
#endif
}
__device__ __forceinline__ void cp_wait_keep4() {
#if USE_ASYNC
  __builtin_amdgcn_s_wait_asynccnt(4);   // allow next tile's 4 ops in flight
#endif
}

// ---------------------------------------------------------------------------
// Per-lane fragment gather matching CDNA5 16-bit A-matrix 16x32 VGPR layout:
//   lane L: row = L%16, half = L/16
//   vgpr p (packed pair): k = (p>=4 ? 16 : 0) + half*8 + (p&3)*2
// Folds into two contiguous ds_load_b128 per fragment.
// B fragments use the same gather on tiles stored [n][k] (pre-transposed).
// ---------------------------------------------------------------------------
__device__ __forceinline__ v16bf load_frag(const bf16_t* __restrict__ base,
                                           int pitch, int mlane, int half) {
  const bf16_t* row = base + mlane * pitch;
  v16bf f;
#pragma unroll
  for (int p = 0; p < 8; ++p) {
    int k = ((p & 4) << 2) + (half << 3) + ((p & 3) << 1);
    f[2 * p]     = row[k];
    f[2 * p + 1] = row[k + 1];
  }
  return f;
}

// ---------------------------------------------------------------------------
// Elementwise / layout helpers
// ---------------------------------------------------------------------------
__global__ void k_cast_bf16(const float* __restrict__ in, bf16_t* __restrict__ out,
                            long n) {
  long i = (long)blockIdx.x * blockDim.x + threadIdx.x;
  if (i < n) out[i] = (bf16_t)in[i];
}

// out[C][R] = bf16(in[R][C]) : LDS 32x32 tile transpose
__global__ __launch_bounds__(256)
void k_cast_transpose(const float* __restrict__ in, bf16_t* __restrict__ out,
                      int R, int C) {
  __shared__ float t[32][33];
  int c0 = blockIdx.x * 32, r0 = blockIdx.y * 32;
  int tx = threadIdx.x & 31, ty = threadIdx.x >> 5;
#pragma unroll
  for (int i = 0; i < 4; ++i)
    t[ty + i * 8][tx] = in[(long)(r0 + ty + i * 8) * C + c0 + tx];
  __syncthreads();
#pragma unroll
  for (int i = 0; i < 4; ++i)
    out[(long)(c0 + ty + i * 8) * R + r0 + tx] = (bf16_t)t[tx][ty + i * 8];
}

// vT[(b*h)][dim][s] = bf16(vf[(b*s+j)][h*64+dim])
__global__ __launch_bounds__(256)
void k_vT(const float* __restrict__ vf, bf16_t* __restrict__ vT,
          int s, int heads) {
  __shared__ float t[32][33];
  int bh = blockIdx.y >> 1;
  int d0 = (blockIdx.y & 1) * 32;
  int b = bh / heads, h = bh % heads;
  int j0 = blockIdx.x * 32;
  int tx = threadIdx.x & 31, ty = threadIdx.x >> 5;
#pragma unroll
  for (int i = 0; i < 4; ++i)
    t[ty + i * 8][tx] = vf[((long)b * s + j0 + ty + i * 8) * (heads * 64) + h * 64 + d0 + tx];
  __syncthreads();
#pragma unroll
  for (int i = 0; i < 4; ++i)
    vT[((long)bh * 64 + d0 + ty + i * 8) * s + j0 + tx] = (bf16_t)t[tx][ty + i * 8];
}

__global__ __launch_bounds__(256)
void k_rmsnorm_cast(const float* __restrict__ x, const float* __restrict__ g,
                    bf16_t* __restrict__ out, int d) {
  __shared__ float red[8];
  long row = blockIdx.x;
  const float* xr = x + row * d;
  float ss = 0.f;
  for (int c = threadIdx.x; c < d; c += 256) { float v = xr[c]; ss += v * v; }
#pragma unroll
  for (int o = 16; o > 0; o >>= 1) ss += __shfl_xor(ss, o, 32);
  if ((threadIdx.x & 31) == 0) red[threadIdx.x >> 5] = ss;
  __syncthreads();
  if (threadIdx.x == 0) {
    float t = 0.f;
#pragma unroll
    for (int i = 0; i < 8; ++i) t += red[i];
    red[0] = rsqrtf(t / (float)d + 1e-5f);
  }
  __syncthreads();
  float rinv = red[0];
  bf16_t* orow = out + row * d;
  for (int c = threadIdx.x; c < d; c += 256)
    orow[c] = (bf16_t)(xr[c] * rinv * g[c]);
}

__global__ void k_rope_cast(const float* __restrict__ in, bf16_t* __restrict__ out,
                            int s, int heads, int hd, long npairs) {
  long idx = (long)blockIdx.x * blockDim.x + threadIdx.x;
  if (idx >= npairs) return;
  int hp = hd >> 1;
  long t   = idx;
  int  i   = (int)(t % hp); t /= hp;
  int  h   = (int)(t % heads); t /= heads;
  long row = t;
  int  pos = (int)(row % s);
  float freq = __expf(-((2.f * i) / (float)hd) * 9.2103403719762f); // ln(10000)
  float ang = (float)pos * freq;
  float c, sn;
  __sincosf(ang, &sn, &c);
  long base = row * (long)heads * hd + (long)h * hd + 2 * i;
  float xe = in[base], xo = in[base + 1];
  out[base]     = (bf16_t)(xe * c - xo * sn);
  out[base + 1] = (bf16_t)(xe * sn + xo * c);
}

// ---------------------------------------------------------------------------
// Tiled bf16 WMMA GEMM with pre-transposed B: C[M,N] = A[M,K] * Bt[N,K]^T.
// Block 256 thr = 8 waves; tile 128x128x32; double-buffered LDS; async
// global->LDS staging overlapped with WMMA compute.
// mode 0: outF = acc            mode 1: outF = acc + resF
// mode 2: outB = bf16(silu(acc)*auxB)   mode 3: outB = bf16(acc)
// ---------------------------------------------------------------------------
#define BM 128
#define BN 128
#define BK 32
#define APITCH 40
#define BPITCH 40

__global__ __launch_bounds__(256)
void k_gemm(const bf16_t* __restrict__ A, const bf16_t* __restrict__ Bt,
            int M, int N, int K, int mode,
            const float* __restrict__ resF, const bf16_t* __restrict__ auxB,
            float* __restrict__ outF, bf16_t* __restrict__ outB) {
  __shared__ __align__(16) bf16_t As[2][BM * APITCH];
  __shared__ __align__(16) bf16_t Bs[2][BN * BPITCH];

  const int tid   = threadIdx.x;
  const int lane  = tid & 31;
  const int wave  = tid >> 5;
  const int wm    = wave >> 2;   // 0..1
  const int wn    = wave & 3;    // 0..3
  const int mlane = lane & 15;
  const int half  = lane >> 4;
  const int m0 = blockIdx.y * BM;
  const int n0 = blockIdx.x * BN;

  // per-thread staging coordinates: 512 b128 chunks per tile, 2 per thread
  const int sr0 = tid >> 2;                // rows tid/4 and tid/4+64
  const int sc  = (tid & 3) * 8;           // k-chunk within row

  auto stage = [&](int buf, int k0) {
#pragma unroll
    for (int e = 0; e < 2; ++e) {
      int r = sr0 + e * 64;
      cp_b128(&As[buf][r * APITCH + sc], &A [(long)(m0 + r) * K + k0 + sc]);
      cp_b128(&Bs[buf][r * BPITCH + sc], &Bt[(long)(n0 + r) * K + k0 + sc]);
    }
  };

  v8f acc[4][2];
#pragma unroll
  for (int i = 0; i < 4; ++i)
#pragma unroll
    for (int j = 0; j < 2; ++j) acc[i][j] = (v8f)0.f;

  stage(0, 0);
  const int nk = K / BK;
  for (int it = 0; it < nk; ++it) {
    const int cur = it & 1;
    if (it + 1 < nk) {
      stage(cur ^ 1, (it + 1) * BK);   // issue next tile's async loads early
      cp_wait_keep4();                 // current tile done; next stays in flight
    } else {
      cp_wait_all();
    }
    __syncthreads();

    v16bf bfrag[2];
#pragma unroll
    for (int nt = 0; nt < 2; ++nt)
      bfrag[nt] = load_frag(&Bs[cur][(wn * 32 + nt * 16) * BPITCH], BPITCH, mlane, half);
#pragma unroll
    for (int mt = 0; mt < 4; ++mt) {
      v16bf a = load_frag(&As[cur][(wm * 64 + mt * 16) * APITCH], APITCH, mlane, half);
#pragma unroll
      for (int nt = 0; nt < 2; ++nt)
        acc[mt][nt] = __builtin_amdgcn_wmma_f32_16x16x32_bf16(
            false, a, false, bfrag[nt], (short)0, acc[mt][nt], false, false);
    }
    __syncthreads();
  }

  const int mw = m0 + wm * 64;
  const int nw = n0 + wn * 32;
#pragma unroll
  for (int mt = 0; mt < 4; ++mt)
#pragma unroll
    for (int nt = 0; nt < 2; ++nt)
#pragma unroll
      for (int r = 0; r < 8; ++r) {
        int  row = mw + mt * 16 + half * 8 + r;
        int  col = nw + nt * 16 + mlane;
        long idx = (long)row * N + col;
        float v = acc[mt][nt][r];
        if (mode == 0)      outF[idx] = v;
        else if (mode == 1) outF[idx] = v + resF[idx];
        else if (mode == 2) {
          float sg = v / (1.f + __expf(-v));
          outB[idx] = (bf16_t)(sg * (float)auxB[idx]);
        } else              outB[idx] = (bf16_t)v;
      }
  (void)M;
}

// ---------------------------------------------------------------------------
// Flash attention (causal), hd=64, bf16 WMMA, f32 online softmax.
// Block = 128 thr (4 waves); 64 q rows per block; 32-key tiles via async LDS.
// V is pre-transposed to [(b*h)][hd][s] so all staging is contiguous b128.
// ---------------------------------------------------------------------------
#define QTL 64
#define KTL 32
#define AHD 64
#define QP  72
#define KP  72
#define VP  40
#define PP  40

__global__ __launch_bounds__(128)
void k_attn(const bf16_t* __restrict__ q, const bf16_t* __restrict__ k,
            const bf16_t* __restrict__ vT, bf16_t* __restrict__ ctx,
            int s, int heads) {
  __shared__ __align__(16) bf16_t Qs[QTL * QP];
  __shared__ __align__(16) bf16_t Ks[KTL * KP];   // [key][dim]
  __shared__ __align__(16) bf16_t Vt[AHD * VP];   // [dim][key]
  __shared__ __align__(16) bf16_t Ps[4][16 * PP];

  const int tid   = threadIdx.x;
  const int lane  = tid & 31;
  const int wave  = tid >> 5;
  const int mlane = lane & 15;
  const int half  = lane >> 4;

  const int bh = blockIdx.y;
  const int b  = bh / heads;
  const int h  = bh % heads;
  const int q0 = blockIdx.x * QTL;

  const long rs = (long)heads * AHD;
  const bf16_t* qb  = q  + (long)b * s * rs + (long)h * AHD;
  const bf16_t* kb  = k  + (long)b * s * rs + (long)h * AHD;
  const bf16_t* vTg = vT + (long)bh * AHD * s;

  // Q tile: 64 rows x 64 dims = 512 chunks / 128 thr
#pragma unroll
  for (int e = 0; e < 4; ++e) {
    int idx = e * 128 + tid;
    int r = idx >> 3, c = (idx & 7) * 8;
    cp_b128(&Qs[r * QP + c], &qb[(long)(q0 + r) * rs + c]);
  }
  cp_wait_all();
  __syncthreads();

  v16bf aQ[2];
  aQ[0] = load_frag(&Qs[(wave * 16) * QP],      QP, mlane, half);
  aQ[1] = load_frag(&Qs[(wave * 16) * QP + 32], QP, mlane, half);

  v8f o[4];
#pragma unroll
  for (int i = 0; i < 4; ++i) o[i] = (v8f)0.f;
  float ms[8], ls[8];
#pragma unroll
  for (int r = 0; r < 8; ++r) { ms[r] = -3.0e38f; ls[r] = 0.f; }

  const int qrow0 = q0 + wave * 16 + half * 8;

  for (int j0 = 0; j0 < q0 + QTL; j0 += KTL) {
    __syncthreads();
    // K tile: 32 rows x 64 dims = 256 chunks; Vt tile: 64 rows x 32 keys
#pragma unroll
    for (int e = 0; e < 2; ++e) {
      int idx = e * 128 + tid;
      int kr = idx >> 3, kc = (idx & 7) * 8;
      cp_b128(&Ks[kr * KP + kc], &kb[(long)(j0 + kr) * rs + kc]);
      int vr = idx >> 2, vc = (idx & 3) * 8;
      cp_b128(&Vt[vr * VP + vc], &vTg[(long)vr * s + j0 + vc]);
    }
    cp_wait_all();
    __syncthreads();

    // S(16x32) = Q(16x64) * K^T(64x32)
    v8f sacc[2];
#pragma unroll
    for (int nt = 0; nt < 2; ++nt) {
      sacc[nt] = (v8f)0.f;
#pragma unroll
      for (int kt = 0; kt < 2; ++kt) {
        v16bf bk = load_frag(&Ks[(nt * 16) * KP + kt * 32], KP, mlane, half);
        sacc[nt] = __builtin_amdgcn_wmma_f32_16x16x32_bf16(
            false, aQ[kt], false, bk, (short)0, sacc[nt], false, false);
      }
    }

    float sv[2][8];
#pragma unroll
    for (int nt = 0; nt < 2; ++nt) {
      int kcol = j0 + nt * 16 + mlane;
#pragma unroll
      for (int r = 0; r < 8; ++r) {
        float x = sacc[nt][r] * 0.125f;      // 1/sqrt(64)
        sv[nt][r] = (kcol <= qrow0 + r) ? x : -3.0e38f;
      }
    }

    float pscale[8];
#pragma unroll
    for (int r = 0; r < 8; ++r) {
      float mx = fmaxf(sv[0][r], sv[1][r]);
#pragma unroll
      for (int off = 8; off > 0; off >>= 1) mx = fmaxf(mx, __shfl_xor(mx, off, 32));
      float mnew = fmaxf(ms[r], mx);
      float corr = __expf(ms[r] - mnew);
      float p0 = __expf(sv[0][r] - mnew);
      float p1 = __expf(sv[1][r] - mnew);
      float rsum = p0 + p1;
#pragma unroll
      for (int off = 8; off > 0; off >>= 1) rsum += __shfl_xor(rsum, off, 32);
      ls[r] = ls[r] * corr + rsum;
      ms[r] = mnew;
      pscale[r] = corr;
      sv[0][r] = p0; sv[1][r] = p1;
    }
#pragma unroll
    for (int dt = 0; dt < 4; ++dt)
#pragma unroll
      for (int r = 0; r < 8; ++r) o[dt][r] *= pscale[r];

    bf16_t* Pw = &Ps[wave][0];
#pragma unroll
    for (int nt = 0; nt < 2; ++nt)
#pragma unroll
      for (int r = 0; r < 8; ++r)
        Pw[(half * 8 + r) * PP + nt * 16 + mlane] = (bf16_t)sv[nt][r];

    v16bf aP = load_frag(Pw, PP, mlane, half);

#pragma unroll
    for (int dt = 0; dt < 4; ++dt) {
      v16bf bv = load_frag(&Vt[(dt * 16) * VP], VP, mlane, half);
      o[dt] = __builtin_amdgcn_wmma_f32_16x16x32_bf16(
          false, aP, false, bv, (short)0, o[dt], false, false);
    }
  }

#pragma unroll
  for (int dt = 0; dt < 4; ++dt)
#pragma unroll
    for (int r = 0; r < 8; ++r) {
      int qr = qrow0 + r;
      long idx = ((long)b * s + qr) * rs + (long)h * AHD + dt * 16 + mlane;
      ctx[idx] = (bf16_t)(o[dt][r] / ls[r]);
    }
}

// ---------------------------------------------------------------------------
// Host orchestration
// ---------------------------------------------------------------------------
extern "C" void kernel_launch(void* const* d_in, const int* in_sizes, int n_in,
                              void* d_out, int out_size, void* d_ws, size_t ws_size,
                              hipStream_t stream) {
  (void)n_in; (void)out_size; (void)ws_size;
  const int D = 1024, DFF = 4096, S = 2048, HEADS = 16;
  const long ROWS = (long)in_sizes[0] / D;     // b*s = 8192

  const float* x  = (const float*)d_in[0];
  const float* g1 = (const float*)d_in[1];
  const float* g2 = (const float*)d_in[2];
  const float* Wq = (const float*)d_in[3];
  const float* Wk = (const float*)d_in[4];
  const float* Wv = (const float*)d_in[5];
  const float* Wo = (const float*)d_in[6];
  const float* W1 = (const float*)d_in[7];
  const float* W2 = (const float*)d_in[8];
  const float* W3 = (const float*)d_in[9];
  float* out = (float*)d_out;

  size_t off = 0;
  auto alloc = [&](size_t bytes) {
    void* p = (char*)d_ws + off;
    off += (bytes + 255) & ~(size_t)255;
    return p;
  };
  const size_t DD = (size_t)D * D, DF = (size_t)D * DFF;
  bf16_t* Wqb = (bf16_t*)alloc(DD * 2);   // all weights stored transposed [N][K]
  bf16_t* Wkb = (bf16_t*)alloc(DD * 2);
  bf16_t* Wvb = (bf16_t*)alloc(DD * 2);
  bf16_t* Wob = (bf16_t*)alloc(DD * 2);
  bf16_t* W1b = (bf16_t*)alloc(DF * 2);
  bf16_t* W2b = (bf16_t*)alloc(DF * 2);
  bf16_t* W3b = (bf16_t*)alloc(DF * 2);
  bf16_t* xnb  = (bf16_t*)alloc((size_t)ROWS * D * 2);
  float*  qf   = (float*) alloc((size_t)ROWS * D * 4);
  float*  kf   = (float*) alloc((size_t)ROWS * D * 4);
  float*  vf   = (float*) alloc((size_t)ROWS * D * 4);
  bf16_t* qbm  = (bf16_t*)alloc((size_t)ROWS * D * 2);
  bf16_t* kbm  = (bf16_t*)alloc((size_t)ROWS * D * 2);
  bf16_t* vTb  = (bf16_t*)alloc((size_t)ROWS * D * 2);  // [(b*h)][hd][s]
  bf16_t* ctxb = (bf16_t*)alloc((size_t)ROWS * D * 2);
  float*  y    = (float*) alloc((size_t)ROWS * D * 4);
  bf16_t* ynb  = (bf16_t*)alloc((size_t)ROWS * D * 2);
  bf16_t* h3b  = (bf16_t*)alloc((size_t)ROWS * DFF * 2);
  bf16_t* hbb  = (bf16_t*)alloc((size_t)ROWS * DFF * 2);

  // weights: cast + transpose to [N][K]
  dim3 tb(256);
  k_cast_transpose<<<dim3(D / 32,   D / 32), tb, 0, stream>>>(Wq, Wqb, D, D);
  k_cast_transpose<<<dim3(D / 32,   D / 32), tb, 0, stream>>>(Wk, Wkb, D, D);
  k_cast_transpose<<<dim3(D / 32,   D / 32), tb, 0, stream>>>(Wv, Wvb, D, D);
  k_cast_transpose<<<dim3(D / 32,   D / 32), tb, 0, stream>>>(Wo, Wob, D, D);
  k_cast_transpose<<<dim3(DFF / 32, D / 32), tb, 0, stream>>>(W1, W1b, D, DFF);
  k_cast_transpose<<<dim3(D / 32, DFF / 32), tb, 0, stream>>>(W2, W2b, DFF, D);
  k_cast_transpose<<<dim3(DFF / 32, D / 32), tb, 0, stream>>>(W3, W3b, D, DFF);

  // xn = rmsnorm(x, g1)
  k_rmsnorm_cast<<<(int)ROWS, 256, 0, stream>>>(x, g1, xnb, D);

  dim3 gdd(D / BN, (int)(ROWS / BM));
  dim3 gdf(DFF / BN, (int)(ROWS / BM));

  // Q/K/V projections (f32 out)
  k_gemm<<<gdd, tb, 0, stream>>>(xnb, Wqb, (int)ROWS, D, D, 0, nullptr, nullptr, qf, nullptr);
  k_gemm<<<gdd, tb, 0, stream>>>(xnb, Wkb, (int)ROWS, D, D, 0, nullptr, nullptr, kf, nullptr);
  k_gemm<<<gdd, tb, 0, stream>>>(xnb, Wvb, (int)ROWS, D, D, 0, nullptr, nullptr, vf, nullptr);

  // RoPE(q), RoPE(k); V -> transposed bf16 [(b*h)][hd][s]
  long npairs = ROWS * HEADS * (AHD / 2);
  k_rope_cast<<<(int)((npairs + 255) / 256), 256, 0, stream>>>(qf, qbm, S, HEADS, AHD, npairs);
  k_rope_cast<<<(int)((npairs + 255) / 256), 256, 0, stream>>>(kf, kbm, S, HEADS, AHD, npairs);
  int BH = (int)(ROWS / S) * HEADS;
  k_vT<<<dim3(S / 32, BH * 2), tb, 0, stream>>>(vf, vTb, S, HEADS);

  // causal flash attention
  k_attn<<<dim3(S / QTL, BH), dim3(128), 0, stream>>>(qbm, kbm, vTb, ctxb, S, HEADS);

  // y = x + ctx @ Wo
  k_gemm<<<gdd, tb, 0, stream>>>(ctxb, Wob, (int)ROWS, D, D, 1, x, nullptr, y, nullptr);

  // yn = rmsnorm(y, g2)
  k_rmsnorm_cast<<<(int)ROWS, 256, 0, stream>>>(y, g2, ynb, D);

  // h3 = yn @ W3 (bf16); hb = silu(yn @ W1) * h3 (bf16)
  k_gemm<<<gdf, tb, 0, stream>>>(ynb, W3b, (int)ROWS, DFF, D, 3, nullptr, nullptr, nullptr, h3b);
  k_gemm<<<gdf, tb, 0, stream>>>(ynb, W1b, (int)ROWS, DFF, D, 2, nullptr, h3b, nullptr, hbb);

  // z = y + hb @ W2
  k_gemm<<<gdd, tb, 0, stream>>>(hbb, W2b, (int)ROWS, D, DFF, 1, y, nullptr, out, nullptr);
}
